// TiramisuModelMult_58239756534249
// MI455X (gfx1250) — matrix operand order repaired and checked by reading, not compiled, not run on hardware
//
#include <hip/hip_runtime.h>
#include <hip/hip_bf16.h>
#include <math.h>

// ---------------------------------------------------------------------------
// CDNA5 (gfx1250) WMMA types
// ---------------------------------------------------------------------------
typedef __attribute__((ext_vector_type(16))) __bf16 v16bf;
typedef __attribute__((ext_vector_type(8)))  __bf16 v8bf;
typedef __attribute__((ext_vector_type(8)))  float  v8f;

#define WAVES_PER_BLOCK 8
#define B_BATCH 1024
#define H_DIM   180
#define H_PAD   192          // 180 padded to 32

__device__ __forceinline__ int imin(int a, int b) { return a < b ? a : b; }

__device__ __forceinline__ v8f wmma_bf16(v16bf a, v16bf b, v8f c) {
    return __builtin_amdgcn_wmma_f32_16x16x32_bf16(false, a, false, b,
                                                   (short)0, c, false, false);
}

// A 16x32 bf16 fragment (ISA layout): elems 0..7 = K[k0+8h .. +7],
// elems 8..15 = K[k0+16+8h .. +7].  16B-aligned b128 loads.
__device__ __forceinline__ v16bf load_frag_a(const __bf16* __restrict__ rowp,
                                             int k0, int half) {
    const v8bf lo = *(const v8bf*)(rowp + k0 + 8 * half);
    const v8bf hi = *(const v8bf*)(rowp + k0 + 16 + 8 * half);
    return __builtin_shufflevector(lo, hi, 0,1,2,3,4,5,6,7,8,9,10,11,12,13,14,15);
}
// B 32x16 bf16 fragment: elems 0..7 = K[k0+8h .. +7], elems 8..15 = K[k0+16+8h .. +7].
__device__ __forceinline__ v16bf load_frag_b(const __bf16* __restrict__ colp,
                                             int k0, int half) {
    const v8bf lo = *(const v8bf*)(colp + k0 + 8 * half);
    const v8bf hi = *(const v8bf*)(colp + k0 + 16 + 8 * half);
    return __builtin_shufflevector(lo, hi, 0,1,2,3,4,5,6,7,8,9,10,11,12,13,14,15);
}

// ---------------------------------------------------------------------------
// bf16 GEMM:  C/Cb = act( A[M,Kp]bf16 @ W[N,Kp]bf16^T + b1 + b2 )
//   A row stride lda (mult of 32), W row stride Kp (mult of 32, zero-padded).
//   Kp is fully padded -> no ragged-K tail, unconditional b128 loads only.
//   C  (optional) f32, row stride N (guarded stores).
//   Cb (optional) bf16, row stride ldcb == 32*ceil(N/32); padding cols get 0.
// One wave32 computes a 32x32 tile (2x2 WMMA).  Wave-uniform early exit keeps
// EXEC all-ones at the WMMA (ISA §7.12 requirement).
// ---------------------------------------------------------------------------
__global__ void gemm_bf16_wmma(
    const __bf16* __restrict__ A, int lda,
    const __bf16* __restrict__ W, int Kp,
    const float* __restrict__ b1,
    const float* __restrict__ b2,
    float*   __restrict__ C,
    __bf16*  __restrict__ Cb, int ldcb,
    int M, int N, int act)
{
    const int lane   = threadIdx.x & 31;
    const int wave   = threadIdx.x >> 5;
    const int tilesN = (N + 31) >> 5;
    const int tilesM = (M + 31) >> 5;
    const int tile   = blockIdx.x * WAVES_PER_BLOCK + wave;
    if (tile >= tilesM * tilesN) return;      // wave-uniform
    const int tm = tile / tilesN;
    const int tn = tile % tilesN;

    const int half = lane >> 4;
    const int l16  = lane & 15;

    const int arow0 = tm * 32 + l16;
    const int arow1 = arow0 + 16;
    const __bf16* __restrict__ a0 = A + (long)imin(arow0, M - 1) * (long)lda;
    const __bf16* __restrict__ a1 = A + (long)imin(arow1, M - 1) * (long)lda;

    const int bcol0 = tn * 32 + l16;
    const int bcol1 = bcol0 + 16;
    const __bf16* __restrict__ w0 = W + (long)imin(bcol0, N - 1) * (long)Kp;
    const __bf16* __restrict__ w1 = W + (long)imin(bcol1, N - 1) * (long)Kp;

    v8f acc00 = {0.f,0.f,0.f,0.f,0.f,0.f,0.f,0.f};
    v8f acc01 = acc00, acc10 = acc00, acc11 = acc00;

    for (int k0 = 0; k0 < Kp; k0 += 32) {
        const v16bf af0 = load_frag_a(a0, k0, half);
        const v16bf af1 = load_frag_a(a1, k0, half);
        const v16bf bf0 = load_frag_b(w0, k0, half);
        const v16bf bf1 = load_frag_b(w1, k0, half);
        acc00 = wmma_bf16(af0, bf0, acc00);
        acc01 = wmma_bf16(af0, bf1, acc01);
        acc10 = wmma_bf16(af1, bf0, acc10);
        acc11 = wmma_bf16(af1, bf1, acc11);
    }

    // ---- epilogue: bias + optional ELU + f32 / padded-bf16 stores ----
    const bool bv0 = bcol0 < N;
    const bool bv1 = bcol1 < N;
    const float bias0 = bv0 ? ((b1 ? b1[bcol0] : 0.f) + (b2 ? b2[bcol0] : 0.f)) : 0.f;
    const float bias1 = bv1 ? ((b1 ? b1[bcol1] : 0.f) + (b2 ? b2[bcol1] : 0.f)) : 0.f;

#pragma unroll
    for (int r = 0; r < 8; ++r) {
        const int row0 = tm * 32 + 8 * half + r;        // rows for acc00/acc01
        const int row1 = row0 + 16;                     // rows for acc10/acc11
        float v00 = acc00[r] + bias0, v01 = acc01[r] + bias1;
        float v10 = acc10[r] + bias0, v11 = acc11[r] + bias1;
        if (act) {
            v00 = (v00 > 0.f) ? v00 : expm1f(v00);
            v01 = (v01 > 0.f) ? v01 : expm1f(v01);
            v10 = (v10 > 0.f) ? v10 : expm1f(v10);
            v11 = (v11 > 0.f) ? v11 : expm1f(v11);
        }
        if (row0 < M) {
            if (C) {
                if (bv0) C[(long)row0 * N + bcol0] = v00;
                if (bv1) C[(long)row0 * N + bcol1] = v01;
            }
            if (Cb) {
                Cb[(long)row0 * ldcb + bcol0] = bv0 ? (__bf16)v00 : (__bf16)0.f;
                Cb[(long)row0 * ldcb + bcol1] = bv1 ? (__bf16)v01 : (__bf16)0.f;
            }
        }
        if (row1 < M) {
            if (C) {
                if (bv0) C[(long)row1 * N + bcol0] = v10;
                if (bv1) C[(long)row1 * N + bcol1] = v11;
            }
            if (Cb) {
                Cb[(long)row1 * ldcb + bcol0] = bv0 ? (__bf16)v10 : (__bf16)0.f;
                Cb[(long)row1 * ldcb + bcol1] = bv1 ? (__bf16)v11 : (__bf16)0.f;
            }
        }
    }
}

// ---------------------------------------------------------------------------
// f32 -> bf16 pack with padded (zero-filled) row stride
// ---------------------------------------------------------------------------
__global__ void pack_bf16_kernel(__bf16* __restrict__ dst,
                                 const float* __restrict__ src,
                                 long rows, int K, int Kp)
{
    const long idx = (long)blockIdx.x * blockDim.x + threadIdx.x;
    if (idx >= rows * Kp) return;
    const long row = idx / Kp;
    const int  col = (int)(idx % Kp);
    dst[idx] = (col < K) ? (__bf16)src[row * K + col] : (__bf16)0.f;
}

__global__ void fill0_f32_kernel(float* __restrict__ p, long n) {
    long i = (long)blockIdx.x * blockDim.x + threadIdx.x;
    long stride = (long)gridDim.x * blockDim.x;
    for (; i < n; i += stride) p[i] = 0.f;
}
__global__ void fill0_bf16_kernel(__bf16* __restrict__ p, long n) {
    long i = (long)blockIdx.x * blockDim.x + threadIdx.x;
    long stride = (long)gridDim.x * blockDim.x;
    for (; i < n; i += stride) p[i] = (__bf16)0.f;
}

// ---------------------------------------------------------------------------
// LSTM gate step:  g = XG[r*T + t, :720] + G[r, :720]
//   i,f,gg,o = split(g,4);  c = sig(f)*c + sig(i)*tanh(gg);  h = sig(o)*tanh(c)
// h is written as padded bf16 (stride H_PAD, pad cols zero); c stays f32.
// ---------------------------------------------------------------------------
__device__ __forceinline__ float sigf(float x) { return 1.f / (1.f + expf(-x)); }

__global__ void lstm_gates_kernel(const float* __restrict__ XG,
                                  const float* __restrict__ G,
                                  __bf16* __restrict__ hb,
                                  float*  __restrict__ c,
                                  int rows, int T, int t)
{
    const long idx = (long)blockIdx.x * blockDim.x + threadIdx.x;
    if (idx >= (long)rows * H_PAD) return;
    const int r = (int)(idx / H_PAD);
    const int j = (int)(idx % H_PAD);
    if (j >= H_DIM) { hb[idx] = (__bf16)0.f; return; }   // padding
    const float* gx = XG + ((long)r * T + t) * (4 * H_DIM);
    const float* gh = G  + (long)r * (4 * H_DIM);
    const float gi = sigf (gx[j            ] + gh[j            ]);
    const float gf = sigf (gx[j +     H_DIM] + gh[j +     H_DIM]);
    const float gg = tanhf(gx[j + 2 * H_DIM] + gh[j + 2 * H_DIM]);
    const float go = sigf (gx[j + 3 * H_DIM] + gh[j + 3 * H_DIM]);
    const long cidx = (long)r * H_DIM + j;
    const float cn = gf * c[cidx] + gi * gg;
    c[cidx] = cn;
    hb[idx] = (__bf16)(go * tanhf(cn));
}

// ---------------------------------------------------------------------------
// Build rec-MLP input directly in padded bf16 (rows = B*Gp, stride 416):
//   cols [0,180)   : nodes_h   (hsrc if hFirst else bcast180)
//   cols [180,360) : comps_h   (bcast180 if hFirst else hsrc)
//   cols [360,386) : loops_tensor[b, li[g], :]
//   cols [386,416) : zero padding
// ---------------------------------------------------------------------------
struct LI8 { int v[8]; };

__global__ void build_rec_input_kernel(__bf16* __restrict__ out,        // stride 416
                                       const float*  __restrict__ bcast180,
                                       const __bf16* __restrict__ hsrc, // stride H_PAD
                                       int hFirst,
                                       const float* __restrict__ loops, // (B,16,26)
                                       LI8 li, int Gp)
{
    const long idx = (long)blockIdx.x * blockDim.x + threadIdx.x;
    const long total = (long)B_BATCH * Gp * 416;
    if (idx >= total) return;
    const long row = idx / 416;
    const int  col = (int)(idx % 416);
    const int  b   = (int)(row / Gp);
    const int  g   = (int)(row % Gp);
    float v;
    if (col < 180) {
        v = hFirst ? (float)hsrc[row * H_PAD + col] : bcast180[col];
    } else if (col < 360) {
        const int j = col - 180;
        v = hFirst ? bcast180[j] : (float)hsrc[row * H_PAD + j];
    } else if (col < 386) {
        v = loops[((long)b * 16 + li.v[g]) * 26 + (col - 360)];
    } else {
        v = 0.f;
    }
    out[idx] = (__bf16)v;
}

// ---------------------------------------------------------------------------
// logits -= 1e10 * (1 - mask)   (in place on d_out, written fresh each call)
// ---------------------------------------------------------------------------
__global__ void mask_kernel(float* __restrict__ out,
                            const float* __restrict__ mask, long n)
{
    const long i = (long)blockIdx.x * blockDim.x + threadIdx.x;
    if (i < n) out[i] = out[i] - 1.0e10f * (1.f - mask[i]);
}

// ---------------------------------------------------------------------------
// Host side
// ---------------------------------------------------------------------------
static inline int pad32(int x) { return (x + 31) & ~31; }

static inline void gemmb(hipStream_t s, const __bf16* A, int lda,
                         const __bf16* W, int Kp,
                         const float* b1, const float* b2,
                         float* C, __bf16* Cb, int ldcb,
                         int M, int N, int act)
{
    const long tiles  = (long)((M + 31) / 32) * ((N + 31) / 32);
    const int  blocks = (int)((tiles + WAVES_PER_BLOCK - 1) / WAVES_PER_BLOCK);
    hipLaunchKernelGGL(gemm_bf16_wmma, dim3(blocks), dim3(256), 0, s,
                       A, lda, W, Kp, b1, b2, C, Cb, ldcb, M, N, act);
}

static inline void pack(hipStream_t s, __bf16* dst, const float* src,
                        long rows, int K, int Kp)
{
    const long n = rows * Kp;
    hipLaunchKernelGGL(pack_bf16_kernel, dim3((int)((n + 255) / 256)), dim3(256),
                       0, s, dst, src, rows, K, Kp);
}

static inline void fill0f(hipStream_t s, float* p, long n) {
    int blocks = (int)((n + 255) / 256); if (blocks > 4096) blocks = 4096;
    hipLaunchKernelGGL(fill0_f32_kernel, dim3(blocks), dim3(256), 0, s, p, n);
}
static inline void fill0b(hipStream_t s, __bf16* p, long n) {
    int blocks = (int)((n + 255) / 256); if (blocks > 4096) blocks = 4096;
    hipLaunchKernelGGL(fill0_bf16_kernel, dim3(blocks), dim3(256), 0, s, p, n);
}

extern "C" void kernel_launch(void* const* d_in, const int* in_sizes, int n_in,
                              void* d_out, int out_size, void* d_ws, size_t ws_size,
                              hipStream_t stream)
{
    (void)in_sizes; (void)n_in; (void)out_size; (void)ws_size;

    // --- inputs (setup_inputs dict order, leaves flattened in order) ---
    const float* obs    = (const float*)d_in[0];   // (1024,32,1052)
    const float* loops  = (const float*)d_in[1];   // (1024,16,26)
    const float* amask  = (const float*)d_in[2];   // (1024,56)
    const float* cW0 = (const float*)d_in[3];  const float* cb0 = (const float*)d_in[4];
    const float* cW1 = (const float*)d_in[5];  const float* cb1 = (const float*)d_in[6];
    const float* cW2 = (const float*)d_in[7];  const float* cb2 = (const float*)d_in[8];
    const float* cW3 = (const float*)d_in[9];  const float* cb3 = (const float*)d_in[10];
    const float* rW0 = (const float*)d_in[11]; const float* rb0 = (const float*)d_in[12];
    const float* rW1 = (const float*)d_in[13]; const float* rb1 = (const float*)d_in[14];
    const float* pW0 = (const float*)d_in[15]; const float* pb0 = (const float*)d_in[16];
    const float* pW1 = (const float*)d_in[17]; const float* pb1 = (const float*)d_in[18];
    const float* lW  = (const float*)d_in[19]; const float* lb  = (const float*)d_in[20];
    const float* cWih = (const float*)d_in[21]; const float* cWhh = (const float*)d_in[22];
    const float* cbih = (const float*)d_in[23]; const float* cbhh = (const float*)d_in[24];
    const float* nWih = (const float*)d_in[25]; const float* nWhh = (const float*)d_in[26];
    const float* nbih = (const float*)d_in[27]; const float* nbhh = (const float*)d_in[28];
    const float* no_comps = (const float*)d_in[29];  // (1,180)
    const float* no_nodes = (const float*)d_in[30];  // (1,180)

    // --- padded dims ---
    const int KP_OBS = pad32(1052);   // 1056
    const int KP_600 = pad32(600);    // 608
    const int KP_350 = pad32(350);    // 352
    const int KP_200 = pad32(200);    // 224
    const int KP_180 = H_PAD;         // 192
    const int KP_386 = pad32(386);    // 416

    // --- workspace carve-up (256B-aligned regions) ---
    char* base = (char*)d_ws;
    size_t off = 0;
    auto alloc = [&](size_t bytes) -> void* {
        off = (off + 255) & ~(size_t)255;
        void* p = base + off;
        off += bytes;
        return p;
    };
    // bf16 weight packs
    __bf16* cW0b  = (__bf16*)alloc(600L  * KP_OBS * 2);
    __bf16* cW1b  = (__bf16*)alloc(350L  * KP_600 * 2);
    __bf16* cW2b  = (__bf16*)alloc(200L  * KP_350 * 2);
    __bf16* cW3b  = (__bf16*)alloc(180L  * KP_200 * 2);
    __bf16* rW0b  = (__bf16*)alloc(200L  * KP_386 * 2);
    __bf16* rW1b  = (__bf16*)alloc(180L  * KP_200 * 2);
    __bf16* pW0b  = (__bf16*)alloc(200L  * KP_180 * 2);
    __bf16* pW1b  = (__bf16*)alloc(180L  * KP_200 * 2);
    __bf16* lWb   = (__bf16*)alloc(56L   * KP_180 * 2);
    __bf16* cWihb = (__bf16*)alloc(720L  * KP_180 * 2);
    __bf16* cWhhb = (__bf16*)alloc(720L  * KP_180 * 2);
    __bf16* nWihb = (__bf16*)alloc(720L  * KP_180 * 2);
    __bf16* nWhhb = (__bf16*)alloc(720L  * KP_180 * 2);
    // bf16 activations (padded strides)
    __bf16* obsb   = (__bf16*)alloc(32768L * KP_OBS * 2);
    __bf16* h1b    = (__bf16*)alloc(32768L * KP_600 * 2);
    __bf16* hs2b   = (__bf16*)alloc(32768L * KP_350 * 2);
    __bf16* hs3b   = (__bf16*)alloc(32768L * KP_200 * 2);
    __bf16* embb   = (__bf16*)alloc(32768L * KP_180 * 2);
    __bf16* hb     = (__bf16*)alloc(8192L  * H_PAD  * 2);
    __bf16* in386b = (__bf16*)alloc(8192L  * KP_386 * 2);
    __bf16* t200b  = (__bf16*)alloc(8192L  * KP_200 * 2);
    __bf16* leafOb = (__bf16*)alloc(8192L  * KP_180 * 2);
    __bf16* midOb  = (__bf16*)alloc(4096L  * KP_180 * 2);
    __bf16* progb  = (__bf16*)alloc(1024L  * KP_180 * 2);
    __bf16* predhb = (__bf16*)alloc(1024L  * KP_200 * 2);
    __bf16* predob = (__bf16*)alloc(1024L  * KP_180 * 2);
    // f32 buffers
    float* XG   = (float*)alloc(32768L * 720 * 4);   // reused 3x
    float* G    = (float*)alloc(8192L  * 720 * 4);   // reused 3x
    float* cbuf = (float*)alloc(8192L  * H_DIM * 4); // reused 3x

    float* outF = (float*)d_out;                     // (1024,56)

    // ---- 0. pack weights + obs to padded bf16 (zero-padded K) ----
    pack(stream, cW0b,  cW0,  600,  1052, KP_OBS);
    pack(stream, cW1b,  cW1,  350,  600,  KP_600);
    pack(stream, cW2b,  cW2,  200,  350,  KP_350);
    pack(stream, cW3b,  cW3,  180,  200,  KP_200);
    pack(stream, rW0b,  rW0,  200,  386,  KP_386);
    pack(stream, rW1b,  rW1,  180,  200,  KP_200);
    pack(stream, pW0b,  pW0,  200,  180,  KP_180);
    pack(stream, pW1b,  pW1,  180,  200,  KP_200);
    pack(stream, lWb,   lW,   56,   180,  KP_180);
    pack(stream, cWihb, cWih, 720,  180,  KP_180);
    pack(stream, cWhhb, cWhh, 720,  180,  KP_180);
    pack(stream, nWihb, nWih, 720,  180,  KP_180);
    pack(stream, nWhhb, nWhh, 720,  180,  KP_180);
    pack(stream, obsb,  obs,  32768, 1052, KP_OBS);

    // ---- 1. comp MLP over all 32768 computation rows (bf16-only outputs) ----
    gemmb(stream, obsb, KP_OBS, cW0b, KP_OBS, cb0, nullptr, nullptr, h1b,  KP_600, 32768, 600, 1);
    gemmb(stream, h1b,  KP_600, cW1b, KP_600, cb1, nullptr, nullptr, hs2b, KP_350, 32768, 350, 1);
    gemmb(stream, hs2b, KP_350, cW2b, KP_350, cb2, nullptr, nullptr, hs3b, KP_200, 32768, 200, 1);
    gemmb(stream, hs3b, KP_200, cW3b, KP_200, cb3, nullptr, nullptr, embb, KP_180, 32768, 180, 1);

    // ---- 2. comps LSTM, all 8 leaves batched (rows = b*8+g, T=4) ----
    // comps_emb row (b*32 + g*4 + t) == XG row ((b*8+g)*4 + t): one big GEMM.
    gemmb(stream, embb, KP_180, cWihb, KP_180, cbih, cbhh, XG, nullptr, 0, 32768, 720, 0);
    fill0b(stream, hb,   8192L * H_PAD);
    fill0f(stream, cbuf, 8192L * H_DIM);
    for (int t = 0; t < 4; ++t) {
        gemmb(stream, hb, H_PAD, cWhhb, KP_180, nullptr, nullptr, G, nullptr, 0, 8192, 720, 0);
        hipLaunchKernelGGL(lstm_gates_kernel,
                           dim3((int)((8192L * H_PAD + 255) / 256)), dim3(256), 0, stream,
                           XG, G, hb, cbuf, 8192, 4, t);
    }

    // ---- 3. leaf rec-MLP:  [no_nodes | comps_h | loop] -> 200 -> 180 ----
    {
        LI8 li = {{2, 3, 5, 6, 8, 9, 11, 12}};
        hipLaunchKernelGGL(build_rec_input_kernel,
                           dim3((int)((8192L * 416 + 255) / 256)), dim3(256), 0, stream,
                           in386b, no_nodes, hb, 0, loops, li, 8);
    }
    gemmb(stream, in386b, KP_386, rW0b, KP_386, rb0, nullptr, nullptr, t200b,  KP_200, 8192, 200, 1);
    gemmb(stream, t200b,  KP_200, rW1b, KP_200, rb1, nullptr, nullptr, leafOb, KP_180, 8192, 180, 1);

    // ---- 4. mid-node nodes-LSTM, 4 nodes batched (rows = b*4+n, T=2) ----
    // leafO row (b*8 + 2n + t) == XG row ((b*4+n)*2 + t): one GEMM.
    gemmb(stream, leafOb, KP_180, nWihb, KP_180, nbih, nbhh, XG, nullptr, 0, 8192, 720, 0);
    fill0b(stream, hb,   4096L * H_PAD);
    fill0f(stream, cbuf, 4096L * H_DIM);
    for (int t = 0; t < 2; ++t) {
        gemmb(stream, hb, H_PAD, nWhhb, KP_180, nullptr, nullptr, G, nullptr, 0, 4096, 720, 0);
        hipLaunchKernelGGL(lstm_gates_kernel,
                           dim3((int)((4096L * H_PAD + 255) / 256)), dim3(256), 0, stream,
                           XG, G, hb, cbuf, 4096, 2, t);
    }

    // ---- 5. mid rec-MLP: [nodes_h | no_comps | loop] ----
    {
        LI8 li = {{1, 4, 7, 10, 0, 0, 0, 0}};
        hipLaunchKernelGGL(build_rec_input_kernel,
                           dim3((int)((4096L * 416 + 255) / 256)), dim3(256), 0, stream,
                           in386b, no_comps, hb, 1, loops, li, 4);
    }
    gemmb(stream, in386b, KP_386, rW0b, KP_386, rb0, nullptr, nullptr, t200b, KP_200, 4096, 200, 1);
    gemmb(stream, t200b,  KP_200, rW1b, KP_200, rb1, nullptr, nullptr, midOb, KP_180, 4096, 180, 1);

    // ---- 6. root nodes-LSTM (rows = b, T=4 over the 4 mid nodes) ----
    gemmb(stream, midOb, KP_180, nWihb, KP_180, nbih, nbhh, XG, nullptr, 0, 4096, 720, 0);
    fill0b(stream, hb,   1024L * H_PAD);
    fill0f(stream, cbuf, 1024L * H_DIM);
    for (int t = 0; t < 4; ++t) {
        gemmb(stream, hb, H_PAD, nWhhb, KP_180, nullptr, nullptr, G, nullptr, 0, 1024, 720, 0);
        hipLaunchKernelGGL(lstm_gates_kernel,
                           dim3((int)((1024L * H_PAD + 255) / 256)), dim3(256), 0, stream,
                           XG, G, hb, cbuf, 1024, 4, t);
    }

    // ---- 7. root rec-MLP ----
    {
        LI8 li = {{0, 0, 0, 0, 0, 0, 0, 0}};
        hipLaunchKernelGGL(build_rec_input_kernel,
                           dim3((int)((1024L * 416 + 255) / 256)), dim3(256), 0, stream,
                           in386b, no_comps, hb, 1, loops, li, 1);
    }
    gemmb(stream, in386b, KP_386, rW0b, KP_386, rb0, nullptr, nullptr, t200b, KP_200, 1024, 200, 1);
    gemmb(stream, t200b,  KP_200, rW1b, KP_200, rb1, nullptr, nullptr, progb, KP_180, 1024, 180, 1);

    // ---- 8. pred MLP + logits + mask ----
    gemmb(stream, progb,  KP_180, pW0b, KP_180, pb0, nullptr, nullptr, predhb, KP_200, 1024, 200, 1);
    gemmb(stream, predhb, KP_200, pW1b, KP_200, pb1, nullptr, nullptr, predob, KP_180, 1024, 180, 1);
    gemmb(stream, predob, KP_180, lWb,  KP_180, lb,  nullptr, outF, nullptr, 0, 1024, 56, 0);
    hipLaunchKernelGGL(mask_kernel,
                       dim3((int)((1024L * 56 + 255) / 256)), dim3(256), 0, stream,
                       outF, amask, 1024L * 56);
}
